// Conv_LIF_69277822485152
// MI455X (gfx1250) — compile-verified, matrix operand order checked by reading, not run on hardware
//
#include <hip/hip_runtime.h>

// CDNA5 / gfx1250 fused Conv7x7 + LIF scan.
//  - conv via v_wmma_f32_16x16x32_f16 (Toeplitz-banded B operand, 7 WMMAs per 16x16 tile per step)
//  - LIF state lives in registers in the WMMA D layout, carried across the t-loop
//  - x read once per block tile (with 3-px halo) per step, spikes written once
//  - global_prefetch_b8 for the next frame while computing the current one

typedef __attribute__((ext_vector_type(16))) _Float16     v16h;
typedef __attribute__((ext_vector_type(8)))  float        v8f;
typedef __attribute__((ext_vector_type(8)))  unsigned int v8u;

#define T_STEPS 128
#define HDIM    512
#define WDIM    512
#define TILE    32                      // block tile: 32x32 pixels (4 waves of 16x16)
#define HALO    3
#define LDS_ROWS  (TILE + 2*HALO)       // 38
#define LDS_PITCH 50                    // f16 units; 25 dwords -> gcd(25,64)=1, no bank conflicts
#define LDS_AROWS (LDS_ROWS + 2)        // pad for benign over-reads (masked by zero B band)

__global__ __launch_bounds__(128)
void conv_lif_wmma_kernel(const float* __restrict__ x,
                          const float* __restrict__ Wk,   // [7][7]
                          float* __restrict__ out)
{
    __shared__ _Float16 smem[LDS_AROWS * LDS_PITCH];

    const int tid   = threadIdx.x;
    const int lane  = tid & 31;
    const int wid   = tid >> 5;             // 0..3
    const int tileY = (wid >> 1) << 4;      // {0,16} local
    const int tileX = (wid & 1)  << 4;      // {0,16} local

    const int blockY = blockIdx.y * TILE;
    const int blockX = blockIdx.x * TILE;

    const int n  = lane & 15;               // WMMA B/D column, A row
    const int hi = lane >> 4;               // K/M half selector
    const int m0 = hi << 3;                 // D-row offset for this half-wave

    // ---- Precompute 7 Toeplitz B operands (32x16 f16 each) from kernel rows ----
    // B_dy[k][n] = W[dy][k-n] if 0 <= k-n <= 6 else 0.
    // 16-bit operand layout: element e of lane holds K = (e<8 ? e : e+8) + 8*hi.
    v16h B[7];
#pragma unroll
    for (int dy = 0; dy < 7; ++dy) {
        v16h b;
#pragma unroll
        for (int e = 0; e < 16; ++e) {
            int k = (e < 8 ? e : e + 8) + (hi << 3);
            int d = k - n;
            float wv = (d >= 0 && d <= 6) ? Wk[dy * 7 + d] : 0.0f;
            b[e] = (_Float16)wv;
        }
        B[dy] = b;
    }

    // ---- LIF state in WMMA-D layout: element r of lane = pixel (m0+r, n) ----
    float vS[8], iS[8];
#pragma unroll
    for (int r = 0; r < 8; ++r) { vS[r] = 0.0f; iS[r] = 0.0f; }

    for (int t = 0; t < T_STEPS; ++t) {
        const float* frame = x + (size_t)t * (HDIM * WDIM);

        // ---- Stage halo tile (38x38 f32 -> f16) into LDS ----
        for (int idx = tid; idx < LDS_ROWS * LDS_ROWS; idx += 128) {
            int lr = idx / LDS_ROWS;
            int lc = idx - lr * LDS_ROWS;
            int gy = blockY + lr - HALO;
            int gx = blockX + lc - HALO;
            float v = 0.0f;
            if (gy >= 0 && gy < HDIM && gx >= 0 && gx < WDIM)
                v = frame[gy * WDIM + gx];
            smem[lr * LDS_PITCH + lc] = (_Float16)v;
        }

        // ---- Prefetch next frame's halo rows into cache while we compute ----
        if (t + 1 < T_STEPS) {
            const float* nf = x + (size_t)(t + 1) * (HDIM * WDIM);
            int pr = tid >> 1, ph = tid & 1;             // one row-half per thread
            if (pr < LDS_ROWS) {
                int gy = blockY + pr - HALO;
                int gx = blockX - HALO + ph * 24;        // two 96B-ish chunks cover 38 floats
                if (gy >= 0 && gy < HDIM && gx >= 0 && gx < WDIM)
                    __builtin_prefetch(&nf[gy * WDIM + gx], 0, 3);
            }
        }

        __syncthreads();

        // ---- 7 banded WMMAs: C = sum_dy A_dy x B_dy ----
        // A_dy[m][k] = in[tileY + m + dy][tileX + k] (local LDS coords, halo included).
        // k >= 22 reads garbage columns but B is zero there (band k-n > 6).
        v8f C = {};
#pragma unroll
        for (int dy = 0; dy < 7; ++dy) {
            const _Float16* arow =
                &smem[(tileY + (lane & 15) + dy) * LDS_PITCH + tileX];
            const unsigned int* arow32 = (const unsigned int*)arow;   // dword aligned
            v8u Au;
#pragma unroll
            for (int j = 0; j < 8; ++j) {
                int kd = (j < 4 ? j : j + 4) + (hi << 2);  // dword index = K/2
                Au[j] = arow32[kd];                         // packed f16 pair (K, K+1)
            }
            v16h A = __builtin_bit_cast(v16h, Au);
            C = __builtin_amdgcn_wmma_f32_16x16x32_f16(
                    false, A, false, B[dy], (short)0, C, false, false);
        }

        // ---- LIF update + spike store (state stays in registers) ----
        float* orow = out + ((size_t)t * HDIM + (blockY + tileY + m0)) * WDIM
                          + (blockX + tileX + n);
#pragma unroll
        for (int r = 0; r < 8; ++r) {
            float i_in  = C[r];
            float v_dec = vS[r] + 0.1f * (iS[r] - vS[r]);  // dt/tau_mem = 0.1, v_leak = 0
            float i_dec = 0.9f * iS[r];                    // dt/tau_syn = 0.1
            float z     = (v_dec > 1.0f) ? 1.0f : 0.0f;    // heaviside(v_dec - v_th)
            vS[r] = (1.0f - z) * v_dec;                    // reset to 0 on spike
            iS[r] = i_dec + i_in;
            orow[(size_t)r * WDIM] = z;
        }

        __syncthreads();   // protect smem before next step's staging
    }
}

extern "C" void kernel_launch(void* const* d_in, const int* in_sizes, int n_in,
                              void* d_out, int out_size, void* d_ws, size_t ws_size,
                              hipStream_t stream) {
    (void)in_sizes; (void)n_in; (void)out_size; (void)d_ws; (void)ws_size;
    const float* x  = (const float*)d_in[0];   // [128,1,512,512]
    const float* Wk = (const float*)d_in[1];   // [1,1,7,7]
    float* out      = (float*)d_out;           // [128,1,512,512]
    dim3 grid(WDIM / TILE, HDIM / TILE);       // 16 x 16 blocks
    conv_lif_wmma_kernel<<<grid, 128, 0, stream>>>(x, Wk, out);
}
